// LSTM_48679159332953
// MI455X (gfx1250) — matrix operand order repaired and checked by reading, not compiled, not run on hardware
//
#include <hip/hip_runtime.h>
#include <hip/hip_bf16.h>

#define T_SEQ   25
#define BATCHN  4096
#define INPUTN  500
#define KPAD    512
#define HID     64
#define GATES   256
#define LAYERS  25
#define OUTN    10
#define TB      (T_SEQ * BATCHN)   // 102400

typedef __attribute__((ext_vector_type(16))) __bf16 v16bf;
typedef __attribute__((ext_vector_type(8)))  __bf16 v8bf;
typedef __attribute__((ext_vector_type(8)))  float  v8f;

// ---------------------------------------------------------------------------
// WMMA fragment loaders (CDNA5 wave32 register layouts, CDNA5 ISA)
// A 16x32 bf16: lane L: m=L&15, half=L>>4:
//   elems 0..7  = K[ 8*half .. 8*half+7 ]
//   elems 8..15 = K[ 16+8*half .. 16+8*half+7 ]
// B 32x16 bf16 (B = W^T, W row-major [N,K]): lane L: n=L&15, half=L>>4:
//   elems 0..7  = K[ 8*half .. 8*half+7 ]
//   elems 8..15 = K[ 16+8*half .. 16+8*half+7 ]  (same as A)
// ---------------------------------------------------------------------------
static __device__ __forceinline__ v16bf load_A_frag(const __bf16* __restrict__ Arow0,
                                                    int lda, int k0, int lane) {
  const int m = lane & 15, half = (lane >> 4) & 1;
  const __bf16* p = Arow0 + (size_t)m * lda + k0;
  v8bf lo = *(const v8bf*)(p + 8 * half);
  v8bf hi = *(const v8bf*)(p + 16 + 8 * half);
  return __builtin_shufflevector(lo, hi, 0, 1, 2, 3, 4, 5, 6, 7,
                                         8, 9, 10, 11, 12, 13, 14, 15);
}

static __device__ __forceinline__ v16bf load_B_frag(const __bf16* __restrict__ W,
                                                    int ldw, int n0, int k0, int lane) {
  const int n = lane & 15, half = (lane >> 4) & 1;
  const __bf16* p = W + (size_t)(n0 + n) * ldw + k0;
  v8bf lo = *(const v8bf*)(p + 8 * half);
  v8bf hi = *(const v8bf*)(p + 16 + 8 * half);
  return __builtin_shufflevector(lo, hi, 0, 1, 2, 3, 4, 5, 6, 7,
                                         8, 9, 10, 11, 12, 13, 14, 15);
}

static __device__ __forceinline__ float sigmoidf_fast(float x) {
  return 1.0f / (1.0f + __expf(-x));
}
static __device__ __forceinline__ float tanhf_fast(float x) {
  float e2 = __expf(2.0f * x);              // overflow-safe: e2=inf -> 1, e2=0 -> -1
  return 1.0f - 2.0f / (e2 + 1.0f);
}

// ---------------------------------------------------------------------------
// Convert f32 -> bf16 with optional K padding (Kp >= K, pad with zeros)
// ---------------------------------------------------------------------------
__global__ void __launch_bounds__(256)
k_cvt_pad(const float* __restrict__ src, __bf16* __restrict__ dst,
          int rows, int K, int Kp) {
  size_t i = (size_t)blockIdx.x * blockDim.x + threadIdx.x;
  size_t total = (size_t)rows * Kp;
  if (i >= total) return;
  int k = (int)(i % Kp);
  size_t r = i / Kp;
  dst[i] = (k < K) ? (__bf16)src[r * (size_t)K + k] : (__bf16)0.0f;
}

// ---------------------------------------------------------------------------
// Input projection: Y_sw = A[M,K](bf16) @ W[256,K]^T(bf16) + bih + bhh
// Output is SWIZZLED to WMMA D-fragment order:
//   Y_sw[ (MT*16 + NT)*256 + lane*8 + r ]  where MT = row/16, NT = col/16,
//   lane = (col&15) + 16*((row>>3)&1), r = row&7.
// Wave tile: 32(M) x 64(N). Block = 4 waves -> 128 M rows. grid.y covers N.
// ---------------------------------------------------------------------------
__global__ void __launch_bounds__(128)
k_inproj(const __bf16* __restrict__ A, int lda, int K,
         const __bf16* __restrict__ W, int ldw,
         const float* __restrict__ bih, const float* __restrict__ bhh,
         float* __restrict__ Y_sw) {
  const int lane = threadIdx.x & 31;
  const int wave = threadIdx.x >> 5;
  const int m0 = (blockIdx.x * 4 + wave) * 32;
  const int n0 = blockIdx.y * 64;

  v8f acc[2][4];
#pragma unroll
  for (int i = 0; i < 2; ++i)
#pragma unroll
    for (int j = 0; j < 4; ++j)
#pragma unroll
      for (int r = 0; r < 8; ++r) acc[i][j][r] = 0.0f;

  for (int k0 = 0; k0 < K; k0 += 32) {
    v16bf a0 = load_A_frag(A + (size_t)m0 * lda, lda, k0, lane);
    v16bf a1 = load_A_frag(A + (size_t)(m0 + 16) * lda, lda, k0, lane);
#pragma unroll
    for (int j = 0; j < 4; ++j) {
      v16bf b = load_B_frag(W, ldw, n0 + j * 16, k0, lane);
      acc[0][j] = __builtin_amdgcn_wmma_f32_16x16x32_bf16(false, a0, false, b,
                                                          (short)0, acc[0][j], false, false);
      acc[1][j] = __builtin_amdgcn_wmma_f32_16x16x32_bf16(false, a1, false, b,
                                                          (short)0, acc[1][j], false, false);
    }
  }

  const int nn = lane & 15;
#pragma unroll
  for (int i = 0; i < 2; ++i) {
    const int MT = (m0 >> 4) + i;
#pragma unroll
    for (int j = 0; j < 4; ++j) {
      const int NT = (n0 >> 4) + j;
      const float bias = bih[NT * 16 + nn] + bhh[NT * 16 + nn];
      v8f v = acc[i][j];
#pragma unroll
      for (int r = 0; r < 8; ++r) v[r] += bias;
      *(v8f*)(Y_sw + ((size_t)(MT * 16 + NT)) * 256 + lane * 8) = v;
    }
  }
}

// ---------------------------------------------------------------------------
// Fused full-layer LSTM recurrence: one launch does all T timesteps.
// Each wave permanently owns 16 batch rows: c-state lives in registers in the
// WMMA D-layout slots; all 32 Whh B-fragments are register-resident; the
// h (D-layout) -> A-layout transpose between steps goes through a per-wave
// private LDS tile. No cross-block communication exists (recurrence is
// elementwise in batch), so no grid sync is needed.
// ---------------------------------------------------------------------------
__global__ void __launch_bounds__(128)
k_lstm_layer(const float* __restrict__ xp_sw,    // swizzled [TB/16][16][32][8]
             const __bf16* __restrict__ Whh,     // [256, 64] bf16
             __bf16* __restrict__ hs_out,        // [T, B, 64] bf16
             float* __restrict__ hN_out,         // [B, 64] f32 (h_n slice)
             float* __restrict__ cN_out) {       // [B, 64] f32 (c_n slice)
  __shared__ __bf16 lds_h[4][16][HID + 8];       // +8 bf16 row pad (anti-conflict)
  const int lane = threadIdx.x & 31;
  const int wave = threadIdx.x >> 5;
  const int m0 = (blockIdx.x * 4 + wave) * 16;   // batch row base for this wave
  const int nn = lane & 15, half = (lane >> 4) & 1;

  // Register-resident Whh fragments: 16 col-tiles x 2 K-halves.
  v16bf Bf[16][2];
#pragma unroll
  for (int nt = 0; nt < 16; ++nt) {
    Bf[nt][0] = load_B_frag(Whh, HID, nt * 16, 0, lane);
    Bf[nt][1] = load_B_frag(Whh, HID, nt * 16, 32, lane);
  }

  v16bf a0 = {}, a1 = {};                        // h_{-1} = 0
  float c[4][8];
#pragma unroll
  for (int q = 0; q < 4; ++q)
#pragma unroll
    for (int r = 0; r < 8; ++r) c[q][r] = 0.0f;

#pragma unroll 1
  for (int t = 0; t < T_SEQ; ++t) {
    const float* xp_t = xp_sw + ((size_t)((t * BATCHN + m0) >> 4)) * (16 * 256);

#pragma unroll
    for (int q = 0; q < 4; ++q) {                // hidden slice q*16 .. q*16+15
      v8f g[4];                                  // gate groups i, f, g, o
#pragma unroll
      for (int s = 0; s < 4; ++s) {
        const int nt = s * 4 + q;
        v8f acc = *(const v8f*)(xp_t + (size_t)nt * 256 + lane * 8);
        acc = __builtin_amdgcn_wmma_f32_16x16x32_bf16(false, a0, false, Bf[nt][0],
                                                      (short)0, acc, false, false);
        acc = __builtin_amdgcn_wmma_f32_16x16x32_bf16(false, a1, false, Bf[nt][1],
                                                      (short)0, acc, false, false);
        g[s] = acc;
      }
      const int hid = q * 16 + nn;
#pragma unroll
      for (int r = 0; r < 8; ++r) {
        const int rowl = r + 8 * half;           // local batch row 0..15
        const float ig = sigmoidf_fast(g[0][r]);
        const float fg = sigmoidf_fast(g[1][r]);
        const float gt = tanhf_fast(g[2][r]);
        const float og = sigmoidf_fast(g[3][r]);
        const float cn = fg * c[q][r] + ig * gt;
        const float hn = og * tanhf_fast(cn);
        c[q][r] = cn;
        lds_h[wave][rowl][hid] = (__bf16)hn;
        if (t == T_SEQ - 1) {
          const size_t idx = (size_t)(m0 + rowl) * HID + hid;
          hN_out[idx] = hn;
          cN_out[idx] = cn;
        }
      }
    }
    __syncthreads();

    // Vectorized writeback of h to global hs (next layer's GEMM input).
#pragma unroll
    for (int p4 = 0; p4 < 4; ++p4) {
      const int flat = p4 * 32 + lane;           // 0..127
      const int row = flat >> 3;                 // 0..15
      const int cg = flat & 7;                   // 8-elem col group
      v8bf hv = *(const v8bf*)(&lds_h[wave][row][cg * 8]);
      *(v8bf*)(hs_out + ((size_t)t * BATCHN + m0 + row) * HID + cg * 8) = hv;
    }

    // Next-step A fragments (h_t) from the per-wave LDS tile.
    {
      const __bf16* p = &lds_h[wave][lane & 15][0];
      v8bf lo = *(const v8bf*)(p + 8 * half);
      v8bf hi = *(const v8bf*)(p + 16 + 8 * half);
      a0 = __builtin_shufflevector(lo, hi, 0, 1, 2, 3, 4, 5, 6, 7,
                                           8, 9, 10, 11, 12, 13, 14, 15);
      lo = *(const v8bf*)(p + 32 + 8 * half);
      hi = *(const v8bf*)(p + 48 + 8 * half);
      a1 = __builtin_shufflevector(lo, hi, 0, 1, 2, 3, 4, 5, 6, 7,
                                           8, 9, 10, 11, 12, 13, 14, 15);
    }
    __syncthreads();
  }
}

// ---------------------------------------------------------------------------
// lstm_out[t, o] = hs_last[t, BATCH-1, :] . W_lin[o, :] + b_lin[o]   (25x10)
// ---------------------------------------------------------------------------
__global__ void __launch_bounds__(256)
k_final(const __bf16* __restrict__ hs_last, const float* __restrict__ Wl,
        const float* __restrict__ bl, float* __restrict__ out) {
  int idx = blockIdx.x * blockDim.x + threadIdx.x;
  if (idx >= T_SEQ * OUTN) return;
  int t = idx / OUTN, o = idx % OUTN;
  const __bf16* h = hs_last + ((size_t)t * BATCHN + (BATCHN - 1)) * HID;
  float s = bl[o];
#pragma unroll
  for (int k = 0; k < HID; ++k) s += (float)h[k] * Wl[o * HID + k];
  out[idx] = s;
}

// ---------------------------------------------------------------------------
extern "C" void kernel_launch(void* const* d_in, const int* in_sizes, int n_in,
                              void* d_out, int out_size, void* d_ws, size_t ws_size,
                              hipStream_t stream) {
  (void)in_sizes; (void)n_in; (void)out_size; (void)ws_size;
  const float* x      = (const float*)d_in[0];
  const float* W_ih0  = (const float*)d_in[1];
  const float* W_hh0  = (const float*)d_in[2];
  const float* b_ih0  = (const float*)d_in[3];
  const float* b_hh0  = (const float*)d_in[4];
  const float* W_ih_r = (const float*)d_in[5];
  const float* W_hh_r = (const float*)d_in[6];
  const float* b_ih_r = (const float*)d_in[7];
  const float* b_hh_r = (const float*)d_in[8];
  const float* W_lin  = (const float*)d_in[9];
  const float* b_lin  = (const float*)d_in[10];

  float* out      = (float*)d_out;
  float* lstm_out = out;                                  // [25, 10]
  float* h_n      = out + T_SEQ * OUTN;                   // [25, 4096, 64]
  float* c_n      = h_n + (size_t)LAYERS * BATCHN * HID;  // [25, 4096, 64]

  char* base = (char*)d_ws;
  size_t off = 0;
  auto carve = [&](size_t bytes) -> char* {
    char* p = base + off;
    off += (bytes + 255) & ~(size_t)255;
    return p;
  };
  __bf16* x_bf  = (__bf16*)carve((size_t)TB * KPAD * 2);              // 104.9 MB
  float*  xp    = (float*) carve((size_t)TB * GATES * 4);             // 104.9 MB
  __bf16* hsA   = (__bf16*)carve((size_t)TB * HID * 2);               // 13.1 MB
  __bf16* hsB   = (__bf16*)carve((size_t)TB * HID * 2);               // 13.1 MB
  __bf16* Wih0b = (__bf16*)carve((size_t)GATES * KPAD * 2);
  __bf16* Whhb  = (__bf16*)carve((size_t)LAYERS * GATES * HID * 2);
  __bf16* Wihrb = (__bf16*)carve((size_t)(LAYERS - 1) * GATES * HID * 2);

  // ---- one-time conversions to bf16 (x padded K 500 -> 512) ----
  {
    size_t n = (size_t)TB * KPAD;
    k_cvt_pad<<<dim3((unsigned)((n + 255) / 256)), 256, 0, stream>>>(x, x_bf, TB, INPUTN, KPAD);
  }
  k_cvt_pad<<<(GATES * KPAD + 255) / 256, 256, 0, stream>>>(W_ih0, Wih0b, GATES, INPUTN, KPAD);
  k_cvt_pad<<<(GATES * HID + 255) / 256, 256, 0, stream>>>(W_hh0, Whhb, GATES, HID, HID);
  k_cvt_pad<<<((LAYERS - 1) * GATES * HID + 255) / 256, 256, 0, stream>>>(
      W_hh_r, Whhb + (size_t)GATES * HID, (LAYERS - 1) * GATES, HID, HID);
  k_cvt_pad<<<((LAYERS - 1) * GATES * HID + 255) / 256, 256, 0, stream>>>(
      W_ih_r, Wihrb, (LAYERS - 1) * GATES, HID, HID);

  // ---- layer loop: 2 launches per layer ----
  const __bf16* layer_in = x_bf;
  __bf16* hs_cur = hsA;
  __bf16* hs_alt = hsB;
  for (int l = 0; l < LAYERS; ++l) {
    const __bf16* Wih = (l == 0) ? Wih0b : (Wihrb + (size_t)(l - 1) * GATES * HID);
    const __bf16* Whh = Whhb + (size_t)l * GATES * HID;
    const float* bi = (l == 0) ? b_ih0 : (b_ih_r + (size_t)(l - 1) * GATES);
    const float* bh = (l == 0) ? b_hh0 : (b_hh_r + (size_t)(l - 1) * GATES);
    const int K = (l == 0) ? KPAD : HID;   // lda == ldw == K (padded layer 0)

    dim3 g(TB / 128, GATES / 64);
    k_inproj<<<g, 128, 0, stream>>>(layer_in, K, K, Wih, K, bi, bh, xp);

    k_lstm_layer<<<BATCHN / 64, 128, 0, stream>>>(
        xp, Whh, hs_cur,
        h_n + (size_t)l * BATCHN * HID,
        c_n + (size_t)l * BATCHN * HID);

    layer_in = hs_cur;
    __bf16* tmp = hs_cur; hs_cur = hs_alt; hs_alt = tmp;
  }

  k_final<<<1, 256, 0, stream>>>((const __bf16*)layer_in, W_lin, b_lin, lstm_out);
}
